// PCD_Align_54511724921390
// MI455X (gfx1250) — matrix operand order repaired and checked by reading, not compiled, not run on hardware
//
#include <hip/hip_runtime.h>

// ---------------------------------------------------------------------------
// PCD-Align pipeline for gfx1250 (MI455X): every conv / deconv / deform-conv
// is an implicit GEMM on v_wmma_f32_16x16x32_bf16 (wave32).
// Block = 256 threads = 8 waves; tile = 128 spatial x (16*NT) output channels.
// Each wave: one 16xK A fragment reused across NT WMMAs (NT=6 for 96-channel
// convs -> grid.y==1, A gathered exactly once; NT=3 for the 144-channel conv).
// ---------------------------------------------------------------------------

typedef __attribute__((ext_vector_type(16))) __bf16 v16bf;
typedef __attribute__((ext_vector_type(8)))  float  v8f;

#define NFC 96
#define GDEF 8
#define LRELU_SLOPE 0.1f

#define M_TILE 128     // spatial outputs per block (8 waves x 16)
#define K_CHUNK 32     // WMMA bf16 K
#define AS_STRIDE 40   // ushorts per LDS row (80B -> rows stay 16B aligned)
#define BS_STRIDE 40

struct __align__(16) U4 { unsigned x, y, z, w; };
union BfFrag { U4 q[2]; v16bf v; };

__device__ __forceinline__ unsigned short f2bf(float f) {
    unsigned u = __float_as_uint(f);
    u += 0x7FFFu + ((u >> 16) & 1u);          // round-to-nearest-even
    return (unsigned short)(u >> 16);
}

// ----------------------------- A-tile gathers ------------------------------
// MODE 0: conv 3x3, pad 1, stride s, optional channel concat (xa|xb split).
// MODE 1: transposed conv 3x3 stride 2 pad 1 outpad 1 as a gather conv:
//         contributes only where (oy+ky) and (ox+kx) are odd.
// MODE 2: DCNv1 — offset-driven bilinear sample fused into the gather.
template<int MODE>
__device__ __forceinline__ float gatherA(const float* __restrict__ xa,
                                         const float* __restrict__ xb, int Csplit,
                                         const float* __restrict__ offs,
                                         int b, int oy, int ox, int cin, int tap,
                                         int Cin, int H, int W, int stride) {
    int ky = tap / 3, kx = tap - 3 * (tap / 3);   // tap in [0,9): cheap
    if (MODE == 0) {
        int iy = oy * stride + ky - 1;
        int ix = ox * stride + kx - 1;
        if ((unsigned)iy >= (unsigned)H || (unsigned)ix >= (unsigned)W) return 0.0f;
        if (cin < Csplit)
            return xa[(((long)b * Csplit + cin) * H + iy) * W + ix];
        int c2 = cin - Csplit, C2 = Cin - Csplit;
        return xb[(((long)b * C2 + c2) * H + iy) * W + ix];
    } else if (MODE == 1) {
        int uy = oy + ky, ux = ox + kx;
        if (!(uy & 1) || !(ux & 1)) return 0.0f;    // lhs-dilation parity
        int iy = (uy - 1) >> 1, ix = (ux - 1) >> 1;
        if (iy >= H || ix >= W) return 0.0f;
        return xa[(((long)b * Cin + cin) * H + iy) * W + ix];
    } else {
        int Cg = Cin / GDEF;
        int g  = cin / Cg;
        long hw = (long)H * W;
        long ob = (((long)b * (2 * GDEF * 9)) + g * 18 + tap * 2) * hw
                  + (long)oy * W + ox;
        float dy = offs[ob];
        float dx = offs[ob + hw];
        float py = (float)(oy + ky - 1) + dy;
        float px = (float)(ox + kx - 1) + dx;
        float y0f = floorf(py), x0f = floorf(px);
        int   y0 = (int)y0f,   x0 = (int)x0f;
        float ly = py - y0f,   lx = px - x0f;
        const float* xc = xa + (((long)b * Cin + cin) * hw);
        float v = 0.0f;
#pragma unroll
        for (int dyi = 0; dyi < 2; ++dyi) {
            int yi = y0 + dyi;
            if ((unsigned)yi >= (unsigned)H) continue;
            float wy = dyi ? ly : 1.0f - ly;
#pragma unroll
            for (int dxi = 0; dxi < 2; ++dxi) {
                int xi = x0 + dxi;
                if ((unsigned)xi >= (unsigned)W) continue;
                float wx = dxi ? lx : 1.0f - lx;
                v += xc[(long)yi * W + xi] * wy * wx;
            }
        }
        return v;
    }
}

template<int MODE>
__device__ __forceinline__ float gatherW(const float* __restrict__ w,
                                         int n, int cin, int tap, int Cin, int Cout) {
    if (MODE == 1) {
        // ConvTranspose weight [Cin,Cout,3,3]; effective OIHW weight is
        // w[cin, n, 2-ky, 2-kx]
        int ky = tap / 3, kx = tap - 3 * (tap / 3);
        return w[(((long)cin * Cout + n) * 3 + (2 - ky)) * 3 + (2 - kx)];
    }
    return w[((long)n * Cin + cin) * 9 + tap];   // OIHW flattened over (cin,tap)
}

// ------------------------- implicit-GEMM WMMA kernel -----------------------
// NT = number of 16-wide N tiles per block (output channels = 16*NT).
template<int MODE, int NT>
__global__ __launch_bounds__(256)
void conv_wmma_kernel(const float* __restrict__ xa, const float* __restrict__ xb,
                      int Csplit,
                      const float* __restrict__ w, const float* __restrict__ bias,
                      const float* __restrict__ offs, float* __restrict__ out,
                      int B, int Cin, int H, int W, int Cout, int Ho, int Wo,
                      int stride, int act) {
    __shared__ __align__(16) unsigned short As[M_TILE * AS_STRIDE];
    __shared__ __align__(16) unsigned short Bs[(NT * 16) * BS_STRIDE];

    const int tid  = threadIdx.x;
    const int lane = tid & 31;
    const int wave = tid >> 5;                 // 8 waves, wave32
    const int mBase = blockIdx.x * M_TILE;
    const int nBase = blockIdx.y * (16 * NT);

    // ---- hoisted per-thread staging coordinates (constant across K loop) ---
    // A staging: this thread always writes As[8*i + wave][lane], i = 0..15.
    int pB[16], pOy[16], pOx[16];
#pragma unroll
    for (int i = 0; i < 16; ++i) {
        int m = mBase + 8 * i + wave;
        int ox = m % Wo; int t = m / Wo;
        pOx[i] = ox; pOy[i] = t % Ho; pB[i] = t / Ho;
    }
    // K decomposition for column k = k0 + lane, maintained incrementally:
    // k += 32  =>  cin += 3, tap += 5 (carry at 9).
    int cinA = lane / 9;
    int tapA = lane - 9 * cinA;

    v8f acc[NT];
#pragma unroll
    for (int j = 0; j < NT; ++j) acc[j] = {};

    const int Ktot = Cin * 9;                  // multiple of 32 for Cin in {96,192}
    for (int k0 = 0; k0 < Ktot; k0 += K_CHUNK) {
        // --- stage A tile (128 x 32) -> LDS as bf16 ---
#pragma unroll
        for (int i = 0; i < 16; ++i) {
            float v = gatherA<MODE>(xa, xb, Csplit, offs, pB[i], pOy[i], pOx[i],
                                    cinA, tapA, Cin, H, W, stride);
            As[(8 * i + wave) * AS_STRIDE + lane] = f2bf(v);
        }
        // --- stage B tile (16*NT x 32), stored [n][k] -> LDS as bf16 ---
#pragma unroll
        for (int i = 0; i < 2 * NT; ++i) {
            int nl = 8 * i + wave;
            float v = gatherW<MODE>(w, nBase + nl, cinA, tapA, Cin, Cout);
            Bs[nl * BS_STRIDE + lane] = f2bf(v);
        }
        // advance (cin, tap) for next chunk (k += 32)
        cinA += 3; tapA += 5;
        if (tapA >= 9) { tapA -= 9; cinA += 1; }
        // speculative prefetch of next chunk's weight line (global_prefetch_b8)
        if (MODE != 1 && k0 + K_CHUNK < Ktot)
            __builtin_prefetch(&w[(long)(nBase + wave) * Ktot + k0 + K_CHUNK], 0, 3);
        __syncthreads();

        // --- assemble fragments per ISA 7.12.2 bf16 layouts (two b128 DS
        //     reads per fragment); 1 A fragment feeds NT WMMAs ---
        BfFrag fa;
        int arow  = wave * 16 + (lane & 15);
        int abase = (lane >> 4) * 8;           // lanes 0-15: K 0-7/16-23; 16-31: 8-15/24-31
        fa.q[0] = *reinterpret_cast<const U4*>(&As[arow * AS_STRIDE + abase]);
        fa.q[1] = *reinterpret_cast<const U4*>(&As[arow * AS_STRIDE + abase + 16]);
        int bcol = lane & 15;
        int kb   = (lane >> 4) * 8;            // lanes 0-15: K 0-7/16-23; 16-31: 8-15/24-31

#pragma unroll
        for (int j = 0; j < NT; ++j) {
            BfFrag fb;
            fb.q[0] = *reinterpret_cast<const U4*>(&Bs[(16 * j + bcol) * BS_STRIDE + kb]);
            fb.q[1] = *reinterpret_cast<const U4*>(&Bs[(16 * j + bcol) * BS_STRIDE + kb + 16]);
            acc[j] = __builtin_amdgcn_wmma_f32_16x16x32_bf16(
                false, fa.v, false, fb.v, (short)0, acc[j], false, false);
        }
        __syncthreads();
    }

    // --- epilogue: bias + optional leaky-ReLU, store 16 x (16*NT) D tile ---
    int col = lane & 15;
#pragma unroll
    for (int j = 0; j < NT; ++j) {
        int n = nBase + 16 * j + col;
        float bv = bias[n];
#pragma unroll
        for (int r = 0; r < 8; ++r) {
            int mrow = r + 8 * (lane >> 4);    // C/D layout: VGPR r -> M=r / r+8
            int m = mBase + wave * 16 + mrow;
            int ox = m % Wo; int t = m / Wo; int oy = t % Ho; int b = t / Ho;
            float v = acc[j][r] + bv;
            if (act) v = (v > 0.0f) ? v : LRELU_SLOPE * v;
            out[(((long)b * Cout + n) * Ho + oy) * Wo + ox] = v;
        }
    }
}

// ----------------------------- elementwise ---------------------------------
__global__ void round_kernel(const float* __restrict__ in, float* __restrict__ out, int n) {
    int i = blockIdx.x * 256 + threadIdx.x;
    if (i < n) out[i] = rintf(in[i]);          // RNE == jnp.round forward value
}
__global__ void add_kernel(const float* __restrict__ a, const float* __restrict__ b,
                           float* __restrict__ out, int n) {
    int i = blockIdx.x * 256 + threadIdx.x;
    if (i < n) out[i] = a[i] + b[i];
}

// ------------------------------- launcher ----------------------------------
static void conv_launch(int mode, const float* xa, const float* xb, int Csplit,
                        const float* w, const float* bias, const float* offs,
                        float* out, int B, int Cin, int H, int W, int Cout,
                        int Ho, int Wo, int stride, int act, hipStream_t s) {
    const int nt = (Cout % 96 == 0) ? 6 : 3;        // 96 -> NT=6 (grid.y=1); 144 -> NT=3
    dim3 grid((unsigned)((B * Ho * Wo) / M_TILE), (unsigned)(Cout / (16 * nt)));
    dim3 block(256);
    if (nt == 6) {
        switch (mode) {
        case 0: conv_wmma_kernel<0, 6><<<grid, block, 0, s>>>(xa, xb, Csplit, w, bias,
                     offs, out, B, Cin, H, W, Cout, Ho, Wo, stride, act); break;
        case 1: conv_wmma_kernel<1, 6><<<grid, block, 0, s>>>(xa, xb, Csplit, w, bias,
                     offs, out, B, Cin, H, W, Cout, Ho, Wo, stride, act); break;
        default: conv_wmma_kernel<2, 6><<<grid, block, 0, s>>>(xa, xb, Csplit, w, bias,
                     offs, out, B, Cin, H, W, Cout, Ho, Wo, stride, act); break;
        }
    } else {
        conv_wmma_kernel<0, 3><<<grid, block, 0, s>>>(xa, xb, Csplit, w, bias,
                 offs, out, B, Cin, H, W, Cout, Ho, Wo, stride, act);
    }
}

extern "C" void kernel_launch(void* const* d_in, const int* in_sizes, int n_in,
                              void* d_out, int out_size, void* d_ws, size_t ws_size,
                              hipStream_t stream) {
    (void)in_sizes; (void)n_in; (void)out_size; (void)ws_size;
    const int B = 2, H = 128, W = 128;

    const float* ref_fea = (const float*)d_in[0];
    const float* inp_fea = (const float*)d_in[1];
    const float* oc1_w = (const float*)d_in[2];   const float* oc1_b = (const float*)d_in[3];
    const float* oc3_w = (const float*)d_in[4];   const float* oc3_b = (const float*)d_in[5];
    const float* enc1_w = (const float*)d_in[6];  const float* enc1_b = (const float*)d_in[7];
    const float* enc2_w = (const float*)d_in[8];  const float* enc2_b = (const float*)d_in[9];
    const float* dec1_w = (const float*)d_in[10]; const float* dec1_b = (const float*)d_in[11];
    const float* dec2_w = (const float*)d_in[12]; const float* dec2_b = (const float*)d_in[13];
    const float* coff_w = (const float*)d_in[14]; const float* coff_b = (const float*)d_in[15];
    const float* dcn_w  = (const float*)d_in[16]; const float* dcn_b  = (const float*)d_in[17];
    const float* ref1_w = (const float*)d_in[18]; const float* ref1_b = (const float*)d_in[19];
    const float* ref2_w = (const float*)d_in[20]; const float* ref2_b = (const float*)d_in[21];

    // outputs: [aligned+r | en_offset | q_offset] flat
    float* out0   = (float*)d_out;
    float* en_off = out0 + (size_t)B * NFC * H * W;
    float* q_off  = en_off + (size_t)B * NFC * 32 * 32;

    // workspace carve-up
    float* ws = (float*)d_ws;
    const size_t pF = (size_t)B * NFC * H * W;       // 3,145,728 floats
    float* t0 = ws;        ws += pF;
    float* t1 = ws;        ws += pF;
    float* e1 = ws;        ws += (size_t)B * NFC * 64 * 64;
    float* d1 = ws;        ws += (size_t)B * NFC * 64 * 64;
    float* de = ws;        ws += pF;
    float* off = ws;       ws += (size_t)B * (2 * GDEF * 9) * H * W;
    float* aligned = ws;   ws += pF;
    float* r1 = ws;        ws += pF;
    float* r2 = ws;        ws += pF;

    // ---- MotionEstimation ----
    conv_launch(0, ref_fea, inp_fea, NFC, oc1_w, oc1_b, nullptr, t0,
                B, 2 * NFC, H, W, NFC, H, W, 1, 1, stream);                 // lrelu
    conv_launch(0, t0, nullptr, NFC, oc3_w, oc3_b, nullptr, t1,
                B, NFC, H, W, NFC, H, W, 1, 1, stream);                     // lrelu
    conv_launch(0, t1, nullptr, NFC, enc1_w, enc1_b, nullptr, e1,
                B, NFC, H, W, NFC, 64, 64, 2, 1, stream);                   // s2 + lrelu
    conv_launch(0, e1, nullptr, NFC, enc2_w, enc2_b, nullptr, en_off,
                B, NFC, 64, 64, NFC, 32, 32, 2, 0, stream);                 // s2

    int nq = B * NFC * 32 * 32;
    round_kernel<<<(nq + 255) / 256, 256, 0, stream>>>(en_off, q_off, nq);

    // ---- MotionCompensation ----
    conv_launch(1, q_off, nullptr, NFC, dec1_w, dec1_b, nullptr, d1,
                B, NFC, 32, 32, NFC, 64, 64, 2, 1, stream);                 // deconv + lrelu
    conv_launch(1, d1, nullptr, NFC, dec2_w, dec2_b, nullptr, de,
                B, NFC, 64, 64, NFC, H, W, 2, 0, stream);                   // deconv
    conv_launch(0, de, nullptr, NFC, coff_w, coff_b, nullptr, off,
                B, NFC, H, W, 2 * GDEF * 9, H, W, 1, 0, stream);            // offsets (144 ch)
    conv_launch(2, ref_fea, nullptr, NFC, dcn_w, dcn_b, off, aligned,
                B, NFC, H, W, NFC, H, W, 1, 0, stream);                     // DCN fused
    conv_launch(0, aligned, ref_fea, NFC, ref1_w, ref1_b, nullptr, r1,
                B, 2 * NFC, H, W, NFC, H, W, 1, 1, stream);                 // lrelu
    conv_launch(0, r1, nullptr, NFC, ref2_w, ref2_b, nullptr, r2,
                B, NFC, H, W, NFC, H, W, 1, 1, stream);                     // lrelu

    int na = B * NFC * H * W;
    add_kernel<<<(na + 255) / 256, 256, 0, stream>>>(aligned, r2, out0, na);
}
